// HGT_20512763806292
// MI455X (gfx1250) — compile-verified, matrix operand order checked by reading, not run on hardware
//
#include <hip/hip_runtime.h>
#include <math.h>

#define HID 256
#define HEADS 8
#define DH 32
#define N_AUTHOR 25000
#define N_PAPER 50000
#define N_TOTAL 75000
#define IN_DIM 128
#define N_LAYERS 2

typedef __attribute__((ext_vector_type(16))) __bf16 v16bf;
typedef __attribute__((ext_vector_type(8)))  __bf16 v8bf;
typedef __attribute__((ext_vector_type(8)))  float  v8f;

// ---------------------------------------------------------------------------
// Elementwise helpers
// ---------------------------------------------------------------------------
__global__ void k_fill_f32(float* p, float v, long n) {
  long i = blockIdx.x * (long)blockDim.x + threadIdx.x;
  long stride = gridDim.x * (long)blockDim.x;
  for (; i < n; i += stride) p[i] = v;
}

// convert f32 -> bf16, optionally applying exact GELU first
__global__ void k_f32_to_bf16(__bf16* dst, const float* src, long n, int gelu) {
  long i = blockIdx.x * (long)blockDim.x + threadIdx.x;
  long stride = gridDim.x * (long)blockDim.x;
  for (; i < n; i += stride) {
    float x = src[i];
    if (gelu) x = 0.5f * x * (1.0f + erff(x * 0.70710678118654752f));
    dst[i] = (__bf16)x;
  }
}

// src f32 [B][R][C] -> dst bf16 [B][C][R]
__global__ void k_transpose_f32_to_bf16(__bf16* dst, const float* src, int R, int C, int B) {
  long n = (long)B * R * C;
  long i = blockIdx.x * (long)blockDim.x + threadIdx.x;
  long stride = gridDim.x * (long)blockDim.x;
  for (; i < n; i += stride) {
    long b = i / ((long)R * C);
    long rem = i - b * (long)R * C;
    int r = (int)(rem / C);
    int c = (int)(rem - (long)r * C);
    dst[b * (long)R * C + (long)c * R + r] = (__bf16)src[i];
  }
}

// ---------------------------------------------------------------------------
// WMMA GEMM: C[M, Ntiles*16] = A_bf16[M, K] x Wt_bf16[N, K]^T (+bias, epilogue)
//   mode 0: plain   mode 1: ReLU   mode 2: gated residual  g*val + (1-g)*resid
// blockIdx.y = head index for batched per-head 32x32 relation transforms
// (headACol / headWElems / headCCol are 0 for plain GEMMs).
//
// Each wave processes TWO column tiles (ct0 = wave, ct1 = wave + nwaves) with
// two independent accumulators sharing the A-tile loads: halves A-side vmem
// traffic and gives two interleaved WMMA dependency chains per wave.
//
// VGPR layouts per CDNA5 ISA 7.12.2 (16-bit A 16x32): lane L holds row L&15,
// K-chunk base (L>>4)*8, elements j<8 -> K=kbase+j, j>=8 -> K=16+kbase+(j-8).
// B (KxN) uses the symmetric layout on columns, so with Wt stored N-major
// (row = output column, stride K) both operands load as two 16-byte runs.
// ---------------------------------------------------------------------------
__global__ void __launch_bounds__(256)
k_gemm_bf16(const __bf16* __restrict__ A, int lda,
            const __bf16* __restrict__ Wt, int K,
            const float* __restrict__ bias,
            float* __restrict__ C, int ldc,
            int M, int Ntiles,
            int mode, const float* __restrict__ resid,
            const float* __restrict__ skipParam,
            int headACol, int headWElems, int headCCol)
{
  int h = blockIdx.y;
  int aCol0 = h * headACol;
  const __bf16* Wp = Wt + (long)h * headWElems;
  int cCol0 = h * headCCol;

  int lane = threadIdx.x & 31;
  int wave = threadIdx.x >> 5;
  int nwaves = blockDim.x >> 5;
  int tileM = blockIdx.x;
  int rown = lane & 15;
  int grp  = lane >> 4;
  int row  = tileM * 16 + rown;
  int rowc = row < M ? row : (M - 1);
  int kbase = grp * 8;
  const __bf16* arow = A + (long)rowc * lda + aCol0;

  float g = 1.0f, og = 0.0f;
  if (mode == 2) {
    float s = skipParam[0];
    g = 1.0f / (1.0f + expf(-s));
    og = 1.0f - g;
  }

  for (int ct0 = wave; ct0 < Ntiles; ct0 += 2 * nwaves) {
    int ct1 = ct0 + nwaves;
    bool has1 = ct1 < Ntiles;                       // wave-uniform
    const __bf16* wrow0 = Wp + (long)(ct0 * 16 + rown) * K;
    const __bf16* wrow1 = has1 ? (Wp + (long)(ct1 * 16 + rown) * K) : wrow0;
    v8f acc0 = {0.f, 0.f, 0.f, 0.f, 0.f, 0.f, 0.f, 0.f};
    v8f acc1 = {0.f, 0.f, 0.f, 0.f, 0.f, 0.f, 0.f, 0.f};
    for (int kc = 0; kc < K; kc += 32) {
      v8bf alo = *(const v8bf*)(arow + kc + kbase);
      v8bf ahi = *(const v8bf*)(arow + kc + 16 + kbase);
      v16bf av = __builtin_shufflevector(alo, ahi, 0,1,2,3,4,5,6,7,8,9,10,11,12,13,14,15);
      v8bf b0lo = *(const v8bf*)(wrow0 + kc + kbase);
      v8bf b0hi = *(const v8bf*)(wrow0 + kc + 16 + kbase);
      v16bf bv0 = __builtin_shufflevector(b0lo, b0hi, 0,1,2,3,4,5,6,7,8,9,10,11,12,13,14,15);
      v8bf b1lo = *(const v8bf*)(wrow1 + kc + kbase);
      v8bf b1hi = *(const v8bf*)(wrow1 + kc + 16 + kbase);
      v16bf bv1 = __builtin_shufflevector(b1lo, b1hi, 0,1,2,3,4,5,6,7,8,9,10,11,12,13,14,15);
      acc0 = __builtin_amdgcn_wmma_f32_16x16x32_bf16(
          false, av, false, bv0, (short)0, acc0, false, false);
      acc1 = __builtin_amdgcn_wmma_f32_16x16x32_bf16(
          false, av, false, bv1, (short)0, acc1, false, false);
    }
    // epilogue: tile ct0
    {
      int colOut = cCol0 + ct0 * 16 + rown;
      float b = bias ? bias[colOut] : 0.0f;
#pragma unroll
      for (int r = 0; r < 8; ++r) {
        int mrow = tileM * 16 + r + grp * 8;        // C/D layout: VGPR r, group grp
        if (mrow < M) {
          float val = acc0[r] + b;
          if (mode == 1) val = fmaxf(val, 0.0f);
          else if (mode == 2) val = g * val + og * resid[(long)mrow * ldc + colOut];
          C[(long)mrow * ldc + colOut] = val;
        }
      }
    }
    // epilogue: tile ct1
    if (has1) {
      int colOut = cCol0 + ct1 * 16 + rown;
      float b = bias ? bias[colOut] : 0.0f;
#pragma unroll
      for (int r = 0; r < 8; ++r) {
        int mrow = tileM * 16 + r + grp * 8;
        if (mrow < M) {
          float val = acc1[r] + b;
          if (mode == 1) val = fmaxf(val, 0.0f);
          else if (mode == 2) val = g * val + og * resid[(long)mrow * ldc + colOut];
          C[(long)mrow * ldc + colOut] = val;
        }
      }
    }
  }
}

// ---------------------------------------------------------------------------
// Edge-phase kernels (fp32, atomics)
// ---------------------------------------------------------------------------
__global__ void k_edge_logits(const int* __restrict__ esrc, const int* __restrict__ edst,
                              const float* __restrict__ q, const float* __restrict__ kt,
                              const float* __restrict__ prel, float scale,
                              float* __restrict__ alpha, int E)
{
  int idx = blockIdx.x * blockDim.x + threadIdx.x;
  if (idx >= E * HEADS) return;
  int e = idx >> 3, h = idx & 7;
  const float* qv = q  + (long)edst[e] * HID + h * DH;
  const float* kv = kt + (long)esrc[e] * HID + h * DH;
  float s = 0.0f;
#pragma unroll
  for (int d = 0; d < DH; d += 4) {
    float4 qa = *(const float4*)(qv + d);
    float4 ka = *(const float4*)(kv + d);
    s += qa.x * ka.x + qa.y * ka.y + qa.z * ka.z + qa.w * ka.w;
  }
  alpha[idx] = s * prel[h] * scale;
}

__device__ inline void atomicMaxF(float* addr, float val) {
  unsigned int* ua = (unsigned int*)addr;
  unsigned int cur = __float_as_uint(*addr);
  while (__uint_as_float(cur) < val) {
    unsigned int prev = atomicCAS(ua, cur, __float_as_uint(val));
    if (prev == cur) break;
    cur = prev;
  }
}

__global__ void k_seg_max(const int* __restrict__ edst, const float* __restrict__ alpha,
                          float* __restrict__ m, int E)
{
  int idx = blockIdx.x * blockDim.x + threadIdx.x;
  if (idx >= E * HEADS) return;
  int e = idx >> 3, h = idx & 7;
  atomicMaxF(&m[(long)edst[e] * HEADS + h], alpha[idx]);
}

__global__ void k_edge_exp(const int* __restrict__ edst, float* __restrict__ alpha,
                           const float* __restrict__ m, float* __restrict__ denom, int E)
{
  int idx = blockIdx.x * blockDim.x + threadIdx.x;
  if (idx >= E * HEADS) return;
  int e = idx >> 3, h = idx & 7;
  long seg = (long)edst[e] * HEADS + h;
  float ev = expf(alpha[idx] - m[seg]);
  alpha[idx] = ev;
  atomicAdd(&denom[seg], ev);
}

__global__ void k_aggregate(const int* __restrict__ esrc, const int* __restrict__ edst,
                            const float* __restrict__ vt, const float* __restrict__ eexp,
                            const float* __restrict__ denom, float* __restrict__ agg, int E)
{
  int idx = blockIdx.x * blockDim.x + threadIdx.x;
  if (idx >= E * HEADS) return;
  int e = idx >> 3, h = idx & 7;
  int d0 = edst[e];
  float att = eexp[idx] / (denom[(long)d0 * HEADS + h] + 1e-16f);
  const float* vv = vt + (long)esrc[e] * HID + h * DH;
  float* out = agg + (long)d0 * HID + h * DH;
#pragma unroll
  for (int d = 0; d < DH; ++d) atomicAdd(&out[d], vv[d] * att);
}

// ---------------------------------------------------------------------------
// Host orchestration
// ---------------------------------------------------------------------------
static inline int idiv(long a, long b) { return (int)((a + b - 1) / b); }
static inline size_t alignup(size_t x) { return (x + 255) & ~(size_t)255; }

extern "C" void kernel_launch(void* const* d_in, const int* in_sizes, int n_in,
                              void* d_out, int out_size, void* d_ws, size_t ws_size,
                              hipStream_t stream) {
  (void)in_sizes; (void)n_in; (void)out_size; (void)ws_size;
  const int TB = 256;

  // --- input index map (JAX tree-flatten: sorted dict keys, depth-first) ---
  // 0 x_author, 1 x_paper, 2 ei_writes, 3 ei_cites, 4 ei_rev_writes,
  // then params: layers[0], layers[1], lin_in
  // per layer (base = 5 + 27*L):
  //  +0  a.author.b  +1  a.author.w  +2  a.paper.b  +3  a.paper.w
  //  +4  a_rel.cites +5  a_rel.rev_writes +6 a_rel.writes
  //  +7  k.author.b  +8  k.author.w  +9  k.paper.b  +10 k.paper.w
  //  +11 m_rel.cites +12 m_rel.rev_writes +13 m_rel.writes
  //  +14 p_rel.cites +15 p_rel.rev_writes +16 p_rel.writes
  //  +17 q.author.b  +18 q.author.w  +19 q.paper.b  +20 q.paper.w
  //  +21 skip.author +22 skip.paper
  //  +23 v.author.b  +24 v.author.w  +25 v.paper.b  +26 v.paper.w
  // lin_in (base 59): 59 author.b, 60 author.w, 61 paper.b, 62 paper.w
  const float* x_author = (const float*)d_in[0];
  const float* x_paper  = (const float*)d_in[1];

  // --- workspace carve-up ---
  char* w = (char*)d_ws;
  size_t off = 0;
  auto carve = [&](size_t bytes) -> void* { void* p = w + off; off = alignup(off + bytes); return p; };

  __bf16* xb   = (__bf16*)carve((size_t)N_TOTAL * HID * 2);  // bf16 activations (also input bf16)
  float*  kbuf = (float*) carve((size_t)N_TOTAL * HID * 4);
  float*  qbuf = (float*) carve((size_t)N_TOTAL * HID * 4);
  float*  vbuf = (float*) carve((size_t)N_TOTAL * HID * 4);
  __bf16* kb   = (__bf16*)carve((size_t)N_TOTAL * HID * 2);
  __bf16* vb   = (__bf16*)carve((size_t)N_TOTAL * HID * 2);
  float*  ktb  = (float*) carve((size_t)N_PAPER * HID * 4);
  float*  vtb  = (float*) carve((size_t)N_PAPER * HID * 4);
  float*  agg  = (float*) carve((size_t)N_TOTAL * HID * 4);
  float*  alpha= (float*) carve((size_t)300000 * HEADS * 4);
  float*  mbuf = (float*) carve((size_t)N_PAPER * HEADS * 4);
  float*  dbuf = (float*) carve((size_t)N_PAPER * HEADS * 4);

  __bf16* lin_wt[2];                 // [author, paper]  [HID][IN_DIM]
  __bf16* proj_wt[N_LAYERS][4][2];   // [layer][k,q,v,a][author,paper]  [HID][HID]
  __bf16* arel_wt[N_LAYERS][3];      // [layer][writes,cites,rev_writes] [H][32][32] (e,d)
  __bf16* mrel_wt[N_LAYERS][3];
  lin_wt[0] = (__bf16*)carve((size_t)HID * IN_DIM * 2);
  lin_wt[1] = (__bf16*)carve((size_t)HID * IN_DIM * 2);
  for (int L = 0; L < N_LAYERS; ++L)
    for (int p = 0; p < 4; ++p)
      for (int t = 0; t < 2; ++t)
        proj_wt[L][p][t] = (__bf16*)carve((size_t)HID * HID * 2);
  for (int L = 0; L < N_LAYERS; ++L)
    for (int r = 0; r < 3; ++r) {
      arel_wt[L][r] = (__bf16*)carve((size_t)HEADS * DH * DH * 2);
      mrel_wt[L][r] = (__bf16*)carve((size_t)HEADS * DH * DH * 2);
    }

  float* xact = (float*)d_out;  // author rows [0,25000), paper rows [25000,75000)
  const float scale = 0.17677669529663687f;  // 1/sqrt(32)

  // --- weight transpose + convert to bf16 ---
  {
    long n = (long)IN_DIM * HID;
    k_transpose_f32_to_bf16<<<idiv(n, TB), TB, 0, stream>>>(lin_wt[0], (const float*)d_in[60], IN_DIM, HID, 1);
    k_transpose_f32_to_bf16<<<idiv(n, TB), TB, 0, stream>>>(lin_wt[1], (const float*)d_in[62], IN_DIM, HID, 1);
  }
  // proj weight indices within a layer: k->(+8,+10) q->(+18,+20) v->(+24,+26) a->(+1,+3)
  const int projW[4][2] = {{8, 10}, {18, 20}, {24, 26}, {1, 3}};
  const int projB[4][2] = {{7,  9}, {17, 19}, {23, 25}, {0, 2}};
  const int arelIdx[3] = {6, 4, 5};    // writes, cites, rev_writes
  const int mrelIdx[3] = {13, 11, 12};
  const int prelIdx[3] = {16, 14, 15};
  for (int L = 0; L < N_LAYERS; ++L) {
    int base = 5 + 27 * L;
    long nh = (long)HID * HID;
    for (int p = 0; p < 4; ++p)
      for (int t = 0; t < 2; ++t)
        k_transpose_f32_to_bf16<<<idiv(nh, TB), TB, 0, stream>>>(
            proj_wt[L][p][t], (const float*)d_in[base + projW[p][t]], HID, HID, 1);
    long nr = (long)HEADS * DH * DH;
    for (int r = 0; r < 3; ++r) {
      k_transpose_f32_to_bf16<<<idiv(nr, TB), TB, 0, stream>>>(
          arel_wt[L][r], (const float*)d_in[base + arelIdx[r]], DH, DH, HEADS);
      k_transpose_f32_to_bf16<<<idiv(nr, TB), TB, 0, stream>>>(
          mrel_wt[L][r], (const float*)d_in[base + mrelIdx[r]], DH, DH, HEADS);
    }
  }

  // --- lin_in: x -> ReLU(x @ W + b), written to d_out (activation buffer) ---
  k_f32_to_bf16<<<idiv((long)N_AUTHOR * IN_DIM, TB), TB, 0, stream>>>(xb, x_author, (long)N_AUTHOR * IN_DIM, 0);
  k_f32_to_bf16<<<idiv((long)N_PAPER * IN_DIM, TB), TB, 0, stream>>>(xb + (size_t)N_AUTHOR * IN_DIM, x_paper, (long)N_PAPER * IN_DIM, 0);
  k_gemm_bf16<<<dim3(idiv(N_AUTHOR, 16), 1), 256, 0, stream>>>(
      xb, IN_DIM, lin_wt[0], IN_DIM, (const float*)d_in[59],
      xact, HID, N_AUTHOR, HID / 16, 1, nullptr, nullptr, 0, 0, 0);
  k_gemm_bf16<<<dim3(idiv(N_PAPER, 16), 1), 256, 0, stream>>>(
      xb + (size_t)N_AUTHOR * IN_DIM, IN_DIM, lin_wt[1], IN_DIM, (const float*)d_in[61],
      xact + (size_t)N_AUTHOR * HID, HID, N_PAPER, HID / 16, 1, nullptr, nullptr, 0, 0, 0);

  // relation table: r = 0 writes (author->paper), 1 cites (paper->paper), 2 rev_writes (paper->author)
  const int  relEi[3]   = {2, 3, 4};
  const int  relE[3]    = {200000, 300000, 200000};
  const long srcOff[3]  = {0, N_AUTHOR, N_AUTHOR};
  const int  nSrc[3]    = {N_AUTHOR, N_PAPER, N_PAPER};
  const long dstOff[3]  = {N_AUTHOR, N_AUTHOR, 0};
  const int  nDst[3]    = {N_PAPER, N_PAPER, N_AUTHOR};

  const long nodeOff[2] = {0, N_AUTHOR};
  const int  nodeN[2]   = {N_AUTHOR, N_PAPER};

  for (int L = 0; L < N_LAYERS; ++L) {
    int base = 5 + 27 * L;
    long nAll = (long)N_TOTAL * HID;

    // bf16 copy of current activations
    k_f32_to_bf16<<<idiv(nAll, TB), TB, 0, stream>>>(xb, xact, nAll, 0);

    // K/Q/V projections
    float* projOut[3] = {kbuf, qbuf, vbuf};
    for (int p = 0; p < 3; ++p)
      for (int t = 0; t < 2; ++t)
        k_gemm_bf16<<<dim3(idiv(nodeN[t], 16), 1), 256, 0, stream>>>(
            xb + (size_t)nodeOff[t] * HID, HID, proj_wt[L][p][t], HID,
            (const float*)d_in[base + projB[p][t]],
            projOut[p] + (size_t)nodeOff[t] * HID, HID, nodeN[t], HID / 16,
            0, nullptr, nullptr, 0, 0, 0);

    k_f32_to_bf16<<<idiv(nAll, TB), TB, 0, stream>>>(kb, kbuf, nAll, 0);
    k_f32_to_bf16<<<idiv(nAll, TB), TB, 0, stream>>>(vb, vbuf, nAll, 0);
    k_fill_f32<<<idiv(nAll, TB), TB, 0, stream>>>(agg, 0.0f, nAll);

    for (int r = 0; r < 3; ++r) {
      int E = relE[r];
      const int* esrc = (const int*)d_in[relEi[r]];
      const int* edst = esrc + E;
      // per-head relation transforms: kt = k @ a_rel, vt = v @ m_rel
      // one wave per block: its dual-tile pair covers the 32-wide head exactly
      k_gemm_bf16<<<dim3(idiv(nSrc[r], 16), HEADS), 32, 0, stream>>>(
          kb + (size_t)srcOff[r] * HID, HID, arel_wt[L][r], DH, nullptr,
          ktb, HID, nSrc[r], 2, 0, nullptr, nullptr, DH, DH * DH, DH);
      k_gemm_bf16<<<dim3(idiv(nSrc[r], 16), HEADS), 32, 0, stream>>>(
          vb + (size_t)srcOff[r] * HID, HID, mrel_wt[L][r], DH, nullptr,
          vtb, HID, nSrc[r], 2, 0, nullptr, nullptr, DH, DH * DH, DH);

      long nseg = (long)nDst[r] * HEADS;
      k_fill_f32<<<idiv(nseg, TB), TB, 0, stream>>>(mbuf, -INFINITY, nseg);
      k_fill_f32<<<idiv(nseg, TB), TB, 0, stream>>>(dbuf, 0.0f, nseg);

      int nEH = E * HEADS;
      k_edge_logits<<<idiv(nEH, TB), TB, 0, stream>>>(
          esrc, edst, qbuf + (size_t)dstOff[r] * HID, ktb,
          (const float*)d_in[base + prelIdx[r]], scale, alpha, E);
      k_seg_max<<<idiv(nEH, TB), TB, 0, stream>>>(edst, alpha, mbuf, E);
      k_edge_exp<<<idiv(nEH, TB), TB, 0, stream>>>(edst, alpha, mbuf, dbuf, E);
      k_aggregate<<<idiv(nEH, TB), TB, 0, stream>>>(
          esrc, edst, vtb, alpha, dbuf, agg + (size_t)dstOff[r] * HID, E);
    }

    // GELU(agg) -> bf16, then A-projection with gated residual (in place on xact)
    k_f32_to_bf16<<<idiv(nAll, TB), TB, 0, stream>>>(xb, agg, nAll, 1);
    for (int t = 0; t < 2; ++t)
      k_gemm_bf16<<<dim3(idiv(nodeN[t], 16), 1), 256, 0, stream>>>(
          xb + (size_t)nodeOff[t] * HID, HID, proj_wt[L][3][t], HID,
          (const float*)d_in[base + projB[3][t]],
          xact + (size_t)nodeOff[t] * HID, HID, nodeN[t], HID / 16,
          2, xact + (size_t)nodeOff[t] * HID,
          (const float*)d_in[base + 21 + t], 0, 0, 0);
  }
}
